// SAGENet_64622077936098
// MI455X (gfx1250) — compile-verified, matrix operand order checked by reading, not run on hardware
//
#include <hip/hip_runtime.h>
#include <hip/hip_bf16.h>

#define N_NODES  50000
#define N_EDGES  800000
#define IN_NODE  128
#define IN_EDGE  32
#define HID      96
#define NUM_LAYERS 3
#define N_GRAPHS 64

typedef __bf16 bf16_t;
typedef __attribute__((ext_vector_type(16))) __bf16 v16bf;
typedef __attribute__((ext_vector_type(8)))  __bf16 v8bf;
typedef __attribute__((ext_vector_type(8)))  float  v8f;
typedef unsigned int v4ui __attribute__((ext_vector_type(4)));
typedef int          v4i  __attribute__((ext_vector_type(4)));
typedef int          v8i  __attribute__((ext_vector_type(8)));

// padded LDS leading dimension (in halves): keeps 16B alignment (multiple of 8)
// and breaks the stride-96 bank-conflict pattern (used for weight buffers).
#define KP(k) ((k) + 8)

// --------------------------- TDM availability ------------------------------
#if defined(__has_builtin)
#if __has_builtin(__builtin_amdgcn_tensor_load_to_lds) && \
    __has_builtin(__builtin_amdgcn_s_wait_tensorcnt)
#define USE_TDM 1
#else
#define USE_TDM 0
#endif
#else
#define USE_TDM 0
#endif

#if defined(__has_include)
#if __has_include(<hip/amd_detail/amd_gfx1250_TDM.h>)
#define TDM_6ARG 1   // amdgpu-toolchain (clang-23): 6-arg builtin
#else
#define TDM_6ARG 0   // ROCm 7.2 (clang-22): 5-arg builtin
#endif
#else
#define TDM_6ARG 0
#endif

// ---------------------------------------------------------------------------
// WMMA fragment helpers (CDNA5 wave32, 16x16x32 bf16)
// A-frag (16x32): lane l -> row (l&15); halves 0..7 = K {kb..kb+7},
// halves 8..15 = K {16+kb..23+kb}, kb = (l>=16)?8:0. B-frag mirrors with N.
// ---------------------------------------------------------------------------
__device__ __forceinline__ v16bf load_frag(const bf16_t* lds, int ld, int k0) {
    const int l  = threadIdx.x & 31;
    const int rc = l & 15;
    const int kb = (l >> 4) << 3;
    const v8bf* q = (const v8bf*)(lds + rc * ld + k0 + kb);
    v8bf lo = q[0];
    v8bf hi = q[2];   // +16 halves
    v16bf f;
#pragma unroll
    for (int j = 0; j < 8; ++j) { f[j] = lo[j]; f[j + 8] = hi[j]; }
    return f;
}

__device__ __forceinline__ v8f wmma_bf16(v16bf a, v16bf b, v8f c) {
    return __builtin_amdgcn_wmma_f32_16x16x32_bf16(
        /*neg_a=*/false, a, /*neg_b=*/false, b,
        /*c_mod=*/(short)0, c, /*reuse_a=*/false, /*reuse_b=*/false);
}

#if USE_TDM
// ---------------------------------------------------------------------------
// Issue a TENSOR_LOAD_TO_LDS for a 2D bf16 tile: rows x cols, row-major,
// global stride == cols, written contiguously into LDS at `ldst`.
// D# built per CDNA5 ISA (groups 0/1; groups 2/3 = 0
// for a 2D tile). Issue from one wave only; completion via TENSORcnt.
// ---------------------------------------------------------------------------
__device__ __forceinline__ void tdm_load_tile_bf16(const bf16_t* gsrc,
                                                   bf16_t* ldst,
                                                   int rows, int cols) {
    unsigned lds_off =
        (unsigned)(size_t)(__attribute__((address_space(3))) bf16_t*)ldst;
    unsigned long long ga = (unsigned long long)(size_t)gsrc;

    // group 0: count=1 | lds_addr | global_addr[56:0] | type=2 ("image")
    v4ui g0;
    g0.x = 1u;
    g0.y = lds_off;
    g0.z = (unsigned)(ga & 0xFFFFFFFFull);
    g0.w = (unsigned)((ga >> 32) & 0x01FFFFFFull) | (2u << 30);

    // group 1: workgroup_mask=0, data_size=1 (2B), no barrier/iterate/pad.
    const unsigned td0 = (unsigned)cols;       // tensor_dim0
    const unsigned td1 = 1u << 20;             // tensor_dim1 (generous, no OOB)
    const unsigned st0 = (unsigned)cols;       // tensor_dim0_stride
    v8i g1;
    g1[0] = (int)(1u << 16);                                   // data_size=2B
    g1[1] = (int)((td0 & 0xFFFFu) << 16);                      // td0[15:0]
    g1[2] = (int)(((td0 >> 16) & 0xFFFFu) | ((td1 & 0xFFFFu) << 16));
    g1[3] = (int)(((td1 >> 16) & 0xFFFFu) |
                  (((unsigned)cols & 0xFFFFu) << 16));         // tile_dim0
    g1[4] = (int)((unsigned)rows & 0xFFFFu);                   // tile_dim1
    g1[5] = (int)st0;                                          // stride0 lo32
    g1[6] = 0;                                                 // stride0 hi16
    g1[7] = 0;
    v4i z4 = {0, 0, 0, 0};
#if TDM_6ARG
    v8i z8 = {0, 0, 0, 0, 0, 0, 0, 0};
    __builtin_amdgcn_tensor_load_to_lds(g0, g1, z4, z4, z8, 0);
#else
    __builtin_amdgcn_tensor_load_to_lds(g0, g1, z4, z4, 0);
#endif
}
#endif  // USE_TDM

// ---------------------------------------------------------------------------
// Fused 2-layer MLP:  out = relu(X @ W1 + b1) @ W2 + b2
// Block = 128 threads (4 waves), tile = 64 rows x HID cols.
// sOvl holds W1^T during GEMM1, then the relu'd mid activation (overlay keeps
// static LDS under 64KB for K1=128).
// ---------------------------------------------------------------------------
template <int K1, bool OUT_BF16>
__global__ __launch_bounds__(128) void mlp2_kernel(
    const float* __restrict__ X,  const float* __restrict__ W1, const float* __restrict__ b1,
    const float* __restrict__ W2, const float* __restrict__ b2, void* __restrict__ out, int M) {

    constexpr int MID_ELEMS = 64 * KP(HID);
    constexpr int W1_ELEMS  = HID * KP(K1);
    constexpr int OVL_ELEMS = (W1_ELEMS > MID_ELEMS) ? W1_ELEMS : MID_ELEMS;

    __shared__ __align__(16) bf16_t sOvl[OVL_ELEMS];
    __shared__ __align__(16) bf16_t sW2[HID * KP(HID)];
    __shared__ __align__(16) bf16_t sA[64 * KP(K1)];

    const int tid     = threadIdx.x;
    const int wave    = tid >> 5;
    const int rowBase = blockIdx.x * 64;

    // W1^T -> sOvl, W2^T -> sW2 (global row-major [K][N] -> LDS [N][Kpad])
    for (int idx = tid; idx < K1 * HID; idx += 128) {
        int k = idx / HID, n = idx - k * HID;
        sOvl[n * KP(K1) + k] = (bf16_t)W1[idx];
    }
    for (int idx = tid; idx < HID * HID; idx += 128) {
        int k = idx / HID, n = idx - k * HID;
        sW2[n * KP(HID) + k] = (bf16_t)W2[idx];
    }
    for (int idx = tid; idx < 64 * K1; idx += 128) {
        int r = idx / K1, c = idx - r * K1;
        int g = rowBase + r;
        sA[r * KP(K1) + c] = (bf16_t)((g < M) ? X[(size_t)g * K1 + c] : 0.0f);
    }
    __syncthreads();

    const v8f vzero = {0, 0, 0, 0, 0, 0, 0, 0};
    v8f acc[6];
#pragma unroll
    for (int t = 0; t < 6; ++t) acc[t] = vzero;

    // GEMM1
    const bf16_t* aRow = sA + (wave * 16) * KP(K1);
    for (int k0 = 0; k0 < K1; k0 += 32) {
        v16bf a = load_frag(aRow, KP(K1), k0);
#pragma unroll
        for (int t = 0; t < 6; ++t) {
            v16bf b = load_frag(sOvl + (t * 16) * KP(K1), KP(K1), k0);
            acc[t] = wmma_bf16(a, b, acc[t]);
        }
    }
    __syncthreads();   // everyone done reading W1 before overlay write

    // epilogue 1: relu(acc + b1) -> sOvl as mid activation [m][n]
    {
        const int l = tid & 31, n0 = l & 15, mh = (l >> 4) << 3;
#pragma unroll
        for (int t = 0; t < 6; ++t)
#pragma unroll
            for (int r = 0; r < 8; ++r) {
                int m = wave * 16 + mh + r;
                int n = t * 16 + n0;
                sOvl[m * KP(HID) + n] = (bf16_t)fmaxf(acc[t][r] + b1[n], 0.0f);
            }
    }
    __syncthreads();

    // GEMM2
#pragma unroll
    for (int t = 0; t < 6; ++t) acc[t] = vzero;
    const bf16_t* aRow2 = sOvl + (wave * 16) * KP(HID);
    for (int k0 = 0; k0 < HID; k0 += 32) {
        v16bf a = load_frag(aRow2, KP(HID), k0);
#pragma unroll
        for (int t = 0; t < 6; ++t) {
            v16bf b = load_frag(sW2 + (t * 16) * KP(HID), KP(HID), k0);
            acc[t] = wmma_bf16(a, b, acc[t]);
        }
    }

    // epilogue 2: acc + b2 -> global (f32 or bf16)
    {
        const int l = tid & 31, n0 = l & 15, mh = (l >> 4) << 3;
#pragma unroll
        for (int t = 0; t < 6; ++t)
#pragma unroll
            for (int r = 0; r < 8; ++r) {
                int gm = rowBase + wave * 16 + mh + r;
                if (gm >= M) continue;
                int n = t * 16 + n0;
                float v = acc[t][r] + b2[n];
                if (OUT_BF16) ((bf16_t*)out)[(size_t)gm * HID + n] = (bf16_t)v;
                else          ((float*)out)[(size_t)gm * HID + n]  = v;
            }
    }
}

// ---------------------------------------------------------------------------
// Per-layer edge kernel:  t = ea @ Wt + bt ; msg = t + h[src] ;
// agg[dst] += msg (fused — the 307MB message tensor is never materialized).
// ea tile staged by the Tensor Data Mover when available (pure bf16 copy,
// overlapped with the Wt f32->bf16 conversion done by the ALUs).
// ---------------------------------------------------------------------------
__global__ __launch_bounds__(128) void edge_msg_kernel(
    const bf16_t* __restrict__ ea, const float* __restrict__ Wt, const float* __restrict__ bt,
    const float* __restrict__ h,   const int* __restrict__ src, const int* __restrict__ dst,
    float* __restrict__ agg) {

    __shared__ __align__(16) bf16_t sW[HID * KP(HID)];
    __shared__ __align__(16) bf16_t sE[64 * HID];   // unpadded: TDM writes rows contiguously

    const int tid  = threadIdx.x;
    const int wave = tid >> 5;
    const int base = blockIdx.x * 64;   // N_EDGES % 64 == 0

#if USE_TDM
    if (wave == 0)
        tdm_load_tile_bf16(ea + (size_t)base * HID, sE, 64, HID);
#endif

    for (int idx = tid; idx < HID * HID; idx += 128) {
        int k = idx / HID, n = idx - k * HID;
        sW[n * KP(HID) + k] = (bf16_t)Wt[idx];
    }

#if USE_TDM
    if (wave == 0)
        __builtin_amdgcn_s_wait_tensorcnt(0);
    asm volatile("" ::: "memory");
#else
    for (int idx = tid; idx < 64 * HID; idx += 128) {
        int r = idx / HID, c = idx - r * HID;
        sE[r * HID + c] = ea[(size_t)(base + r) * HID + c];
    }
#endif
    __syncthreads();

    const v8f vzero = {0, 0, 0, 0, 0, 0, 0, 0};
    v8f acc[6];
#pragma unroll
    for (int t = 0; t < 6; ++t) acc[t] = vzero;

    const bf16_t* aRow = sE + (wave * 16) * HID;
    for (int k0 = 0; k0 < HID; k0 += 32) {
        v16bf a = load_frag(aRow, HID, k0);
#pragma unroll
        for (int t = 0; t < 6; ++t) {
            v16bf b = load_frag(sW + (t * 16) * KP(HID), KP(HID), k0);
            acc[t] = wmma_bf16(a, b, acc[t]);
        }
    }

    const int l = tid & 31, n0 = l & 15, mh = (l >> 4) << 3;
#pragma unroll
    for (int r = 0; r < 8; ++r) {
        int e = base + wave * 16 + mh + r;
        int s = src[e], d = dst[e];
        const float* hrow = h   + (size_t)s * HID;
        float*       arow = agg + (size_t)d * HID;
#pragma unroll
        for (int t = 0; t < 6; ++t) {
            int n = t * 16 + n0;
            atomicAdd(&arow[n], acc[t][r] + bt[n] + hrow[n]);
        }
    }
}

// ---------------------------------------------------------------------------
// Node update:  conv = (agg/cnt) @ Wl + bl + h @ Wr ;
// hout = relu(h + conv)  (layers 0..L-2)  or  conv  (last layer).
// Two WMMA passes reuse one LDS weight buffer to stay < 64KB.
// ---------------------------------------------------------------------------
__global__ __launch_bounds__(128) void node_update_kernel(
    const float* __restrict__ h,  const float* __restrict__ agg, const float* __restrict__ cnt,
    const float* __restrict__ Wl, const float* __restrict__ bl,  const float* __restrict__ Wr,
    float* __restrict__ hout, int M, int do_relu) {

    __shared__ __align__(16) bf16_t sW[HID * KP(HID)];
    __shared__ __align__(16) bf16_t sH[64 * KP(HID)];
    __shared__ __align__(16) bf16_t sG[64 * KP(HID)];

    const int tid     = threadIdx.x;
    const int wave    = tid >> 5;
    const int rowBase = blockIdx.x * 64;

    for (int idx = tid; idx < 64 * HID; idx += 128) {
        int r = idx / HID, c = idx - r * HID;
        int g = rowBase + r;
        float hv = 0.0f, gv = 0.0f;
        if (g < M) {
            hv = h[(size_t)g * HID + c];
            gv = agg[(size_t)g * HID + c] / fmaxf(cnt[g], 1.0f);
        }
        sH[r * KP(HID) + c] = (bf16_t)hv;
        sG[r * KP(HID) + c] = (bf16_t)gv;
    }
    for (int idx = tid; idx < HID * HID; idx += 128) {
        int k = idx / HID, n = idx - k * HID;
        sW[n * KP(HID) + k] = (bf16_t)Wl[idx];
    }
    __syncthreads();

    const v8f vzero = {0, 0, 0, 0, 0, 0, 0, 0};
    v8f acc[6];
#pragma unroll
    for (int t = 0; t < 6; ++t) acc[t] = vzero;

    // pass 1: acc += (agg/cnt) @ Wl
    const bf16_t* gRow = sG + (wave * 16) * KP(HID);
    for (int k0 = 0; k0 < HID; k0 += 32) {
        v16bf a = load_frag(gRow, KP(HID), k0);
#pragma unroll
        for (int t = 0; t < 6; ++t) {
            v16bf b = load_frag(sW + (t * 16) * KP(HID), KP(HID), k0);
            acc[t] = wmma_bf16(a, b, acc[t]);
        }
    }
    __syncthreads();
    for (int idx = tid; idx < HID * HID; idx += 128) {
        int k = idx / HID, n = idx - k * HID;
        sW[n * KP(HID) + k] = (bf16_t)Wr[idx];
    }
    __syncthreads();

    // pass 2: acc += h @ Wr
    const bf16_t* hRow = sH + (wave * 16) * KP(HID);
    for (int k0 = 0; k0 < HID; k0 += 32) {
        v16bf a = load_frag(hRow, KP(HID), k0);
#pragma unroll
        for (int t = 0; t < 6; ++t) {
            v16bf b = load_frag(sW + (t * 16) * KP(HID), KP(HID), k0);
            acc[t] = wmma_bf16(a, b, acc[t]);
        }
    }

    const int l = tid & 31, n0 = l & 15, mh = (l >> 4) << 3;
#pragma unroll
    for (int t = 0; t < 6; ++t)
#pragma unroll
        for (int r = 0; r < 8; ++r) {
            int gm = rowBase + wave * 16 + mh + r;
            if (gm >= M) continue;
            int n = t * 16 + n0;
            float conv = acc[t][r] + bl[n];
            float old  = h[(size_t)gm * HID + n];
            hout[(size_t)gm * HID + n] = do_relu ? fmaxf(old + conv, 0.0f) : conv;
        }
}

// ---------------------------------------------------------------------------
// Small helper kernels
// ---------------------------------------------------------------------------
__global__ void count_kernel(const int* __restrict__ dst, float* __restrict__ cnt, int E) {
    int e = blockIdx.x * blockDim.x + threadIdx.x;
    if (e < E) atomicAdd(&cnt[dst[e]], 1.0f);
}

__global__ void pool_acc_kernel(const float* __restrict__ h, const int* __restrict__ batch,
                                float* __restrict__ pooled, float* __restrict__ gcnt, int M) {
    long long idx = (long long)blockIdx.x * blockDim.x + threadIdx.x;
    if (idx < (long long)M * HID) {
        int node = (int)(idx / HID), f = (int)(idx - (long long)node * HID);
        int g = batch[node];
        atomicAdd(&pooled[g * HID + f], h[idx]);
        if (f == 0) atomicAdd(&gcnt[g], 1.0f);
    }
}

__global__ void pool_final_kernel(const float* __restrict__ pooled, const float* __restrict__ gcnt,
                                  float* __restrict__ out) {
    int idx = blockIdx.x * blockDim.x + threadIdx.x;
    if (idx < N_GRAPHS * HID) {
        int g = idx / HID;
        out[idx] = pooled[idx] / fmaxf(gcnt[g], 1.0f);
    }
}

// ---------------------------------------------------------------------------
// Host launcher
// ---------------------------------------------------------------------------
extern "C" void kernel_launch(void* const* d_in, const int* in_sizes, int n_in,
                              void* d_out, int out_size, void* d_ws, size_t ws_size,
                              hipStream_t stream) {
    const float* x    = (const float*)d_in[0];
    const int*   eidx = (const int*)  d_in[1];
    const float* eat  = (const float*)d_in[2];
    const int*   batch= (const int*)  d_in[3];
    const float* Wn1  = (const float*)d_in[4];
    const float* bn1  = (const float*)d_in[5];
    const float* Wn2  = (const float*)d_in[6];
    const float* bn2  = (const float*)d_in[7];
    const float* We1  = (const float*)d_in[8];
    const float* be1  = (const float*)d_in[9];
    const float* We2  = (const float*)d_in[10];
    const float* be2  = (const float*)d_in[11];
    const float* Wt   = (const float*)d_in[12];
    const float* bt   = (const float*)d_in[13];
    const float* Wl   = (const float*)d_in[14];
    const float* bl   = (const float*)d_in[15];
    const float* Wr   = (const float*)d_in[16];

    const int* src = eidx;
    const int* dst = eidx + N_EDGES;

    // workspace carve-up (~212 MB)
    float*  h0     = (float*)d_ws;
    float*  h1     = h0  + (size_t)N_NODES * HID;
    float*  agg    = h1  + (size_t)N_NODES * HID;
    float*  cnt    = agg + (size_t)N_NODES * HID;
    float*  pooled = cnt + N_NODES;
    float*  gcnt   = pooled + N_GRAPHS * HID;
    bf16_t* ea     = (bf16_t*)(gcnt + N_GRAPHS);   // 16B-aligned by construction

    hipMemsetAsync(cnt, 0, N_NODES * sizeof(float), stream);
    hipMemsetAsync(pooled, 0, (N_GRAPHS * HID + N_GRAPHS) * sizeof(float), stream);

    count_kernel<<<(N_EDGES + 255) / 256, 256, 0, stream>>>(dst, cnt, N_EDGES);

    const int nodeBlocks = (N_NODES + 63) / 64;   // 782
    const int edgeBlocks = N_EDGES / 64;          // 12500

    mlp2_kernel<IN_NODE, false><<<nodeBlocks, 128, 0, stream>>>(x, Wn1, bn1, Wn2, bn2, h0, N_NODES);
    mlp2_kernel<IN_EDGE, true ><<<edgeBlocks, 128, 0, stream>>>(eat, We1, be1, We2, be2, ea, N_EDGES);

    float* hin = h0;
    float* ho  = h1;
    for (int i = 0; i < NUM_LAYERS; ++i) {
        hipMemsetAsync(agg, 0, (size_t)N_NODES * HID * sizeof(float), stream);
        edge_msg_kernel<<<edgeBlocks, 128, 0, stream>>>(
            ea, Wt + (size_t)i * HID * HID, bt + (size_t)i * HID, hin, src, dst, agg);
        node_update_kernel<<<nodeBlocks, 128, 0, stream>>>(
            hin, agg, cnt, Wl + (size_t)i * HID * HID, bl + (size_t)i * HID,
            Wr + (size_t)i * HID * HID, ho, N_NODES, (i < NUM_LAYERS - 1) ? 1 : 0);
        float* tmp = hin; hin = ho; ho = tmp;
    }

    long long total = (long long)N_NODES * HID;
    pool_acc_kernel<<<(unsigned)((total + 255) / 256), 256, 0, stream>>>(hin, batch, pooled, gcnt, N_NODES);
    pool_final_kernel<<<(N_GRAPHS * HID + 255) / 256, 256, 0, stream>>>(pooled, gcnt, (float*)d_out);
}
